// SpatioTemporalVisualTokenRefinerModel_30691836297741
// MI455X (gfx1250) — matrix operand order repaired and checked by reading, compiled, not run on hardware
//
#include <hip/hip_runtime.h>
#include <hip/hip_bf16.h>

// ---------------------------------------------------------------------------
// SpatioTemporalVisualTokenRefiner for MI455X (gfx1250, wave32, WMMA)
//   B=2, T=32, N=256, E=1024, H=16, D=64, L=2, K=256, MAXLEN=512
// Compute-bound pipeline: f16 WMMA with f32 accumulate, double-buffered LDS.
// ---------------------------------------------------------------------------

typedef __attribute__((ext_vector_type(16))) _Float16 v16h;
typedef __attribute__((ext_vector_type(8)))  _Float16 v8h;
typedef __attribute__((ext_vector_type(2)))  _Float16 v2h;
typedef __attribute__((ext_vector_type(8)))  float    v8f;
typedef __attribute__((ext_vector_type(4)))  float    v4f;

#define E_DIM   1024
#define H_DIM   16
#define D_DIM   64
#define B_DIM   2
#define T_DIM   32
#define N_DIM   256
#define K_SEL   256
#define M_SEL   8192          // T*N per batch
#define RB_ROWS 1023          // 2*MAXLEN-1

// ---------------------------------------------------------------------------
// WMMA GEMM: C[M,N] = A[M,Kd] * W[Kd,N] (+ bias[N]).
// Block tile 128x128, BK=32, double-buffered LDS, register-prefetched tiles.
// 8 waves: 4 along M x 2 along N -> 32x64 per wave -> 8 WMMA per K-step.
// Requires M%128==0, N%128==0, Kd%32==0 (true for all call sites here).
// ---------------------------------------------------------------------------
__global__ __launch_bounds__(256) void gemm_f16_kernel(
    const float* __restrict__ A, const float* __restrict__ W,
    const float* __restrict__ bias, float* __restrict__ C,
    int M, int N, int Kd)
{
  __shared__ __align__(16) _Float16 As[2][128][40];   // [buf][m][k]
  __shared__ __align__(16) _Float16 Bs[2][128][40];   // [buf][n][k] (transposed)

  const int tid  = threadIdx.x;
  const int lane = tid & 31;
  const int w    = tid >> 5;     // wave id 0..7
  const int wm   = w & 3;        // wave row (M): 32 rows each
  const int wn   = w >> 2;       // wave col (N): 64 cols each
  const int m0   = blockIdx.y * 128;
  const int n0   = blockIdx.x * 128;
  const int half = lane >> 4;
  const int l16  = lane & 15;

  // staging coordinates
  const int ar  = tid >> 1;            // A row 0..127
  const int ac  = (tid & 1) * 16;      // A col chunk (16 floats)
  const int brp = tid >> 4;            // B row pair 0..15 -> k rows 2*brp,2*brp+1
  const int bcg = (tid & 15) * 8;      // B col group (8 cols)

  const float* aptr  = A + (size_t)(m0 + ar) * Kd + ac;
  const float* bptr0 = W + (size_t)(2 * brp) * N + n0 + bcg;

  v4f ra[4], rb0[2], rb1[2];

  auto load_tile = [&](int k0) {
    const float* ap = aptr + k0;
    ra[0] = *(const v4f*)(ap + 0);
    ra[1] = *(const v4f*)(ap + 4);
    ra[2] = *(const v4f*)(ap + 8);
    ra[3] = *(const v4f*)(ap + 12);
    const float* bp = bptr0 + (size_t)k0 * N;
    rb0[0] = *(const v4f*)(bp + 0);
    rb0[1] = *(const v4f*)(bp + 4);
    rb1[0] = *(const v4f*)(bp + N + 0);
    rb1[1] = *(const v4f*)(bp + N + 4);
  };

  auto store_tile = [&](int buf) {
    v8h h0, h1;
    #pragma unroll
    for (int j = 0; j < 4; ++j) {
      h0[j]     = (_Float16)ra[0][j];
      h0[4 + j] = (_Float16)ra[1][j];
      h1[j]     = (_Float16)ra[2][j];
      h1[4 + j] = (_Float16)ra[3][j];
    }
    *(v8h*)&As[buf][ar][ac]     = h0;
    *(v8h*)&As[buf][ar][ac + 8] = h1;
    #pragma unroll
    for (int j = 0; j < 8; ++j) {
      v2h pr;
      pr[0] = (_Float16)(j < 4 ? rb0[0][j] : rb0[1][j - 4]);
      pr[1] = (_Float16)(j < 4 ? rb1[0][j] : rb1[1][j - 4]);
      *(v2h*)&Bs[buf][bcg + j][2 * brp] = pr;
    }
  };

  load_tile(0);
  store_tile(0);
  __syncthreads();

  v8f acc[2][4] = {};
  int cur = 0;

  for (int k0 = 0; k0 < Kd; k0 += 32) {
    const bool hn = (k0 + 32) < Kd;
    if (hn) load_tile(k0 + 32);                    // next tile -> registers
    if (k0 + 64 < Kd) {                            // far prefetch -> L2
      __builtin_prefetch(aptr + k0 + 64, 0, 1);
      __builtin_prefetch(bptr0 + (size_t)(k0 + 64) * N, 0, 1);
    }

    // ---- fragments from current LDS buffer ----
    v16h a[2], b[4];
    #pragma unroll
    for (int mi = 0; mi < 2; ++mi) {
      const _Float16* s = &As[cur][wm * 32 + mi * 16 + l16][half * 8];
      v8h lo = *(const v8h*)s;
      v8h hi = *(const v8h*)(s + 16);
      #pragma unroll
      for (int j = 0; j < 8; ++j) { a[mi][j] = lo[j]; a[mi][8 + j] = hi[j]; }
    }
    #pragma unroll
    for (int ni = 0; ni < 4; ++ni) {
      const _Float16* s = &Bs[cur][wn * 64 + ni * 16 + l16][half * 8];
      v8h lo = *(const v8h*)s;
      v8h hi = *(const v8h*)(s + 16);
      #pragma unroll
      for (int j = 0; j < 8; ++j) { b[ni][j] = lo[j]; b[ni][8 + j] = hi[j]; }
    }

    #pragma unroll
    for (int mi = 0; mi < 2; ++mi)
      #pragma unroll
      for (int ni = 0; ni < 4; ++ni)
        acc[mi][ni] = __builtin_amdgcn_wmma_f32_16x16x32_f16(
            false, a[mi], false, b[ni], (short)0, acc[mi][ni], false, false);

    if (hn) store_tile(cur ^ 1);                   // fill the other buffer
    __syncthreads();
    cur ^= 1;
  }

  // ---- epilogue: bias add + store (C layout: M = r + 8*half, N = l16) ----
  #pragma unroll
  for (int mi = 0; mi < 2; ++mi)
    #pragma unroll
    for (int ni = 0; ni < 4; ++ni) {
      int col = n0 + wn * 64 + ni * 16 + l16;
      float bv = bias ? bias[col] : 0.0f;
      #pragma unroll
      for (int r = 0; r < 8; ++r) {
        int row = m0 + wm * 32 + mi * 16 + half * 8 + r;
        C[(size_t)row * N + col] = acc[mi][ni][r] + bv;
      }
    }
}

// ---------------------------------------------------------------------------
// Fused relative-position attention core for one (sequence, head, 16-row q tile).
// 1 wave per block. scores = QK^T/8 + relbias -> softmax -> ctx = P V.
// Q/K/V/O are [Bx*S, E] with head h in columns h*64 .. h*64+63.
// ---------------------------------------------------------------------------
__global__ __launch_bounds__(32) void attn_kernel(
    const float* __restrict__ Q, const float* __restrict__ K,
    const float* __restrict__ V, float* __restrict__ O,
    const float* __restrict__ rb, int S)
{
  __shared__ __align__(16) float sc[16][N_DIM + 8];  // up to S=256 keys
  __shared__ float rs[16];

  const int lane = threadIdx.x;
  const int half = lane >> 4;
  const int l16  = lane & 15;
  const int q0   = blockIdx.x * 16;
  const int h    = blockIdx.y;
  const int seq  = blockIdx.z;

  const size_t base = (size_t)seq * S * E_DIM + (size_t)h * D_DIM;
  const float* Qp = Q + base;
  const float* Kp = K + base;
  const float* Vp = V + base;
  float*       Op = O + base;

  // ---- Q fragments (16 rows x 64 = two K-steps of 32), vector loads ----
  v16h qa[2];
  #pragma unroll
  for (int ks = 0; ks < 2; ++ks) {
    const float* qs = &Qp[(size_t)(q0 + l16) * E_DIM + ks * 32 + half * 8];
    v4f c0 = *(const v4f*)(qs + 0);
    v4f c1 = *(const v4f*)(qs + 4);
    v4f c2 = *(const v4f*)(qs + 16);
    v4f c3 = *(const v4f*)(qs + 20);
    #pragma unroll
    for (int j = 0; j < 4; ++j) {
      qa[ks][j]      = (_Float16)c0[j];
      qa[ks][4 + j]  = (_Float16)c1[j];
      qa[ks][8 + j]  = (_Float16)c2[j];
      qa[ks][12 + j] = (_Float16)c3[j];
    }
  }

  // ---- scores = Q K^T * (1/sqrt(64)) + rel bias ----
  for (int j0 = 0; j0 < S; j0 += 16) {
    v8f acc = {};
    #pragma unroll
    for (int ks = 0; ks < 2; ++ks) {
      const float* kk = &Kp[(size_t)(j0 + l16) * E_DIM + ks * 32 + half * 8];
      v4f c0 = *(const v4f*)(kk + 0);
      v4f c1 = *(const v4f*)(kk + 4);
      v4f c2 = *(const v4f*)(kk + 16);
      v4f c3 = *(const v4f*)(kk + 20);
      v16h kbf;
      #pragma unroll
      for (int j = 0; j < 4; ++j) {
        kbf[j]      = (_Float16)c0[j];
        kbf[4 + j]  = (_Float16)c1[j];
        kbf[8 + j]  = (_Float16)c2[j];
        kbf[12 + j] = (_Float16)c3[j];
      }
      acc = __builtin_amdgcn_wmma_f32_16x16x32_f16(
          false, qa[ks], false, kbf, (short)0, acc, false, false);
    }
    int t = j0 + l16;
    #pragma unroll
    for (int r = 0; r < 8; ++r) {
      int srow = half * 8 + r;
      int sg   = q0 + srow;
      sc[srow][t] = acc[r] * 0.125f + rb[(size_t)(t - sg + 511) * H_DIM + h];
    }
  }
  __syncthreads();

  // ---- row softmax (2 lanes per row; shfl_xor combine within wave32) ----
  {
    int row = lane >> 1, sub = lane & 1;
    float mx = -3.0e38f;
    for (int c = sub; c < S; c += 2) mx = fmaxf(mx, sc[row][c]);
    mx = fmaxf(mx, __shfl_xor(mx, 1, 32));
    float sum = 0.0f;
    for (int c = sub; c < S; c += 2) {
      float ev = __expf(sc[row][c] - mx);
      sc[row][c] = ev;
      sum += ev;
    }
    sum += __shfl_xor(sum, 1, 32);
    if (sub == 0) rs[row] = sum;
  }
  __syncthreads();

  // ---- ctx = P @ V (P from LDS as f16 A fragments, 32 keys per WMMA) ----
  v8f oacc[4] = {};
  for (int j0 = 0; j0 < S; j0 += 32) {
    const float* ps = &sc[l16][j0 + half * 8];
    v4f c0 = *(const v4f*)(ps + 0);
    v4f c1 = *(const v4f*)(ps + 4);
    v4f c2 = *(const v4f*)(ps + 16);
    v4f c3 = *(const v4f*)(ps + 20);
    v16h p;
    #pragma unroll
    for (int j = 0; j < 4; ++j) {
      p[j]      = (_Float16)c0[j];
      p[4 + j]  = (_Float16)c1[j];
      p[8 + j]  = (_Float16)c2[j];
      p[12 + j] = (_Float16)c3[j];
    }
    #pragma unroll
    for (int dt = 0; dt < 4; ++dt) {
      const float* vs = &Vp[(size_t)(j0 + half * 8) * E_DIM + dt * 16 + l16];
      float vreg[16];
      #pragma unroll
      for (int e = 0; e < 16; ++e) vreg[e] = vs[(size_t)(e + (e >> 3) * 8) * E_DIM];
      v16h vb;
      #pragma unroll
      for (int e = 0; e < 16; ++e) vb[e] = (_Float16)vreg[e];
      oacc[dt] = __builtin_amdgcn_wmma_f32_16x16x32_f16(
          false, p, false, vb, (short)0, oacc[dt], false, false);
    }
  }

  // ---- store, folding 1/rowsum ----
  #pragma unroll
  for (int dt = 0; dt < 4; ++dt)
    #pragma unroll
    for (int r = 0; r < 8; ++r) {
      int row = half * 8 + r;
      Op[(size_t)(q0 + row) * E_DIM + dt * 16 + l16] = oacc[dt][r] / rs[row];
    }
}

// ---------------------------------------------------------------------------
// [B, D1, D2, E] -> [B, D2, D1, E]  (E = 1024, B = 2); contiguous E runs.
// ---------------------------------------------------------------------------
__global__ __launch_bounds__(256) void transpose_kernel(
    const float* __restrict__ in, float* __restrict__ out, int D1, int D2)
{
  size_t gid = (size_t)blockIdx.x * 256 + threadIdx.x;
  size_t total = (size_t)B_DIM * D1 * D2 * E_DIM;
  if (gid >= total) return;
  int e = (int)(gid & (E_DIM - 1));
  size_t r = gid >> 10;
  int d2 = (int)(r % D2); r /= D2;
  int d1 = (int)(r % D1);
  int b  = (int)(r / D1);
  out[(((size_t)b * D2 + d2) * D1 + d1) * E_DIM + e] = in[gid];
}

// ---------------------------------------------------------------------------
// LDS-tiled 2-D transpose: in [B][R][C] -> out [B][C][R]; both sides coalesced.
// ---------------------------------------------------------------------------
__global__ __launch_bounds__(256) void transpose2d_kernel(
    const float* __restrict__ in, float* __restrict__ out, int R, int C)
{
  __shared__ float tile[32][33];
  const int b  = blockIdx.z;
  const int r0 = blockIdx.y * 32;
  const int c0 = blockIdx.x * 32;
  const int tx = threadIdx.x & 31;
  const int ty = threadIdx.x >> 5;           // 8 rows per pass
  const float* ip = in  + (size_t)b * R * C;
  float*       op = out + (size_t)b * R * C;
  #pragma unroll
  for (int i = 0; i < 32; i += 8)
    tile[ty + i][tx] = ip[(size_t)(r0 + ty + i) * C + c0 + tx];
  __syncthreads();
  #pragma unroll
  for (int i = 0; i < 32; i += 8)
    op[(size_t)(c0 + ty + i) * R + r0 + tx] = tile[tx][ty + i];
}

// ---------------------------------------------------------------------------
// Row softmax over M for scT[b][k][:], writing wtsT[b][k][:]; fully coalesced.
// One block per (b*K + k) row.
// ---------------------------------------------------------------------------
__global__ __launch_bounds__(256) void rowsoftmax_kernel(
    const float* __restrict__ scT, float* __restrict__ wtsT, int Mrows)
{
  __shared__ float red[256];
  const int tid = threadIdx.x;
  const float* p = scT  + (size_t)blockIdx.x * Mrows;
  float*       o = wtsT + (size_t)blockIdx.x * Mrows;

  float mx = -3.0e38f;
  for (int m = tid; m < Mrows; m += 256) mx = fmaxf(mx, p[m]);
  red[tid] = mx; __syncthreads();
  for (int s = 128; s > 0; s >>= 1) {
    if (tid < s) red[tid] = fmaxf(red[tid], red[tid + s]);
    __syncthreads();
  }
  mx = red[0]; __syncthreads();

  float sum = 0.0f;
  for (int m = tid; m < Mrows; m += 256) sum += __expf(p[m] - mx);
  red[tid] = sum; __syncthreads();
  for (int s = 128; s > 0; s >>= 1) {
    if (tid < s) red[tid] += red[tid + s];
    __syncthreads();
  }
  float inv = 1.0f / red[0];

  for (int m = tid; m < Mrows; m += 256) o[m] = __expf(p[m] - mx) * inv;
}

// ---------------------------------------------------------------------------
// Multi-scale average pooling: sel [B,256,E] -> out [B, 256+128+64, E]
// ---------------------------------------------------------------------------
__global__ __launch_bounds__(256) void pool_kernel(
    const float* __restrict__ sel, float* __restrict__ out)
{
  size_t gid = (size_t)blockIdx.x * 256 + threadIdx.x;
  const size_t total = (size_t)B_DIM * 448 * E_DIM;
  if (gid >= total) return;
  int e = (int)(gid & (E_DIM - 1));
  size_t r = gid >> 10;
  int row = (int)(r % 448);
  int b   = (int)(r / 448);
  const float* sb = sel + (size_t)b * K_SEL * E_DIM;
  float v;
  if (row < 256) {
    v = sb[(size_t)row * E_DIM + e];
  } else if (row < 384) {
    int rr = (row - 256) * 2;
    v = 0.5f * (sb[(size_t)rr * E_DIM + e] + sb[(size_t)(rr + 1) * E_DIM + e]);
  } else {
    int rr = (row - 384) * 4;
    v = 0.25f * (sb[(size_t)rr * E_DIM + e] + sb[(size_t)(rr + 1) * E_DIM + e] +
                 sb[(size_t)(rr + 2) * E_DIM + e] + sb[(size_t)(rr + 3) * E_DIM + e]);
  }
  out[gid] = v;
}

// ---------------------------------------------------------------------------
// Host-side orchestration
// ---------------------------------------------------------------------------
extern "C" void kernel_launch(void* const* d_in, const int* in_sizes, int n_in,
                              void* d_out, int out_size, void* d_ws, size_t ws_size,
                              hipStream_t stream)
{
  (void)in_sizes; (void)n_in; (void)out_size; (void)ws_size;

  const float* x  = (const float*)d_in[0];
  const float* WQ = (const float*)d_in[1];
  const float* BQ = (const float*)d_in[2];
  const float* WK = (const float*)d_in[3];
  const float* BK = (const float*)d_in[4];
  const float* WV = (const float*)d_in[5];
  const float* BV = (const float*)d_in[6];
  const float* WD = (const float*)d_in[7];
  const float* BD = (const float*)d_in[8];
  const float* RB = (const float*)d_in[9];
  const float* SW = (const float*)d_in[10];
  const float* SB = (const float*)d_in[11];

  const int E = E_DIM, H = H_DIM;
  const size_t HID = (size_t)B_DIM * T_DIM * N_DIM * E;  // 16M floats

  float* ws   = (float*)d_ws;
  float* bufA = ws;                 // hidden state ping
  float* bufB = bufA + HID;         // hidden state pong
  float* qb   = bufB + HID;
  float* kb   = qb + HID;
  float* vb   = kb + HID;
  float* cb   = vb + HID;                               // attention context
  float* scb  = cb + HID;                               // selection scores [B,M,K]
  float* scT  = scb  + (size_t)B_DIM * M_SEL * K_SEL;   // [B,K,M]
  float* wtsT = scT  + (size_t)B_DIM * M_SEL * K_SEL;   // [B,K,M]
  float* selb = wtsT + (size_t)B_DIM * M_SEL * K_SEL;   // [B,K,E]

  auto gemm = [&](const float* A, const float* W, const float* bias, float* C,
                  int M, int Nn, int Kd) {
    dim3 g(Nn / 128, M / 128);
    gemm_f16_kernel<<<g, 256, 0, stream>>>(A, W, bias, C, M, Nn, Kd);
  };

  auto attn = [&](const float* X, float* Y, int a, int Bx, int S) {
    const float* wq = WQ + (size_t)a * E * E;  const float* bq = BQ + (size_t)a * E;
    const float* wk = WK + (size_t)a * E * E;  const float* bk = BK + (size_t)a * E;
    const float* wv = WV + (size_t)a * E * E;  const float* bv = BV + (size_t)a * E;
    const float* wd = WD + (size_t)a * E * E;  const float* bd = BD + (size_t)a * E;
    const float* rb = RB + (size_t)a * RB_ROWS * H;
    const int M = Bx * S;
    gemm(X, wq, bq, qb, M, E, E);
    gemm(X, wk, bk, kb, M, E, E);
    gemm(X, wv, bv, vb, M, E, E);
    dim3 ga(S / 16, H, Bx);
    attn_kernel<<<ga, 32, 0, stream>>>(qb, kb, vb, cb, rb, S);
    gemm(cb, wd, bd, Y, M, E, E);
  };

  const int TBLK = (int)((HID + 255) / 256);

  // ---- layer 0 ----
  attn(x, bufA, 0, B_DIM * T_DIM, N_DIM);                                  // spatial
  transpose_kernel<<<TBLK, 256, 0, stream>>>(bufA, bufB, T_DIM, N_DIM);    // ->[B,N,T,E]
  attn(bufB, bufA, 1, B_DIM * N_DIM, T_DIM);                               // temporal
  transpose_kernel<<<TBLK, 256, 0, stream>>>(bufA, bufB, N_DIM, T_DIM);    // ->[B,T,N,E]

  // ---- layer 1 ----
  attn(bufB, bufA, 2, B_DIM * T_DIM, N_DIM);
  transpose_kernel<<<TBLK, 256, 0, stream>>>(bufA, bufB, T_DIM, N_DIM);
  attn(bufB, bufA, 3, B_DIM * N_DIM, T_DIM);
  transpose_kernel<<<TBLK, 256, 0, stream>>>(bufA, bufB, N_DIM, T_DIM);
  // final hidden state: bufB in [B,T,N,E]

  // ---- differentiable token selection ----
  gemm(bufB, SW, SB, scb, B_DIM * M_SEL, K_SEL, E);                        // scores
  {
    dim3 tg(K_SEL / 32, M_SEL / 32, B_DIM);
    transpose2d_kernel<<<tg, 256, 0, stream>>>(scb, scT, M_SEL, K_SEL);    // ->[B,K,M]
  }
  rowsoftmax_kernel<<<B_DIM * K_SEL, 256, 0, stream>>>(scT, wtsT, M_SEL);
  for (int b = 0; b < B_DIM; ++b) {
    gemm(wtsT + (size_t)b * K_SEL * M_SEL,           // A: [K, M]
         bufB + (size_t)b * M_SEL * E,               // W: [M, E]
         nullptr,
         selb + (size_t)b * K_SEL * E,               // C: [K, E]
         K_SEL, E, M_SEL);
  }

  // ---- multi-scale pooling -> output [B, 448, E] ----
  const int PBLK = (B_DIM * 448 * E_DIM) / 256;
  pool_kernel<<<PBLK, 256, 0, stream>>>(selb, (float*)d_out);
}
